// CapXLayer_45200235823238
// MI455X (gfx1250) — matrix operand order repaired and checked by reading, not compiled, not run on hardware
//
#include <hip/hip_runtime.h>
#include <math.h>

// Problem constants (from reference)
#define ICAP   8
#define IND    16
#define MIDC   32
#define OCAP   8
#define ODIM   16
#define BATCH  8
#define HH     64
#define WW     64
#define NPOS   (HH * WW)          // 4096 positions per image
#define ROUT_ITERS 3

// ---------------- CDNA5 WMMA plumbing ----------------
typedef __attribute__((ext_vector_type(16))) __bf16 v16bf;
typedef __attribute__((ext_vector_type(8)))  __bf16 v8bf;
typedef __attribute__((ext_vector_type(8)))  float  v8f;

union Frag16 { v16bf v; v8bf h8[2]; __bf16 h[16]; };
union FAcc   { v8f   v; float f[8]; };
union Chunk8 { v8bf  v; __bf16 h[8]; };

__device__ __forceinline__ float sigm(float x) { return 1.0f / (1.0f + __expf(-x)); }

// =====================================================================
// Kernel 1: fused grouped conv stack -> u_hat  (WMMA bf16, f32 accum)
// One workgroup = (batch b, group g, 16x16 spatial tile). 256 thr = 8 waves.
// LDS layouts are transposed so WMMA fragments are contiguous 16B chunks:
//   h1  [pix][c]   (B-frag = 2x ds_load_b128)
//   o2  [n][c]     (B-frag = 2x ds_load_b128)
//   w2  [m][K pad 296], w3 [m][K pad 40]  (A-frag halves = ds_load_b128)
// u_hat layout: [b][pos][ic][m=oc*16+od]  (routing-friendly, coalesced)
// =====================================================================
__global__ __launch_bounds__(256)
void conv_stack_kernel(const float* __restrict__ xg,
                       const float* __restrict__ w1g, const float* __restrict__ b1g,
                       const float* __restrict__ w2g, const float* __restrict__ b2g,
                       const float* __restrict__ w3g, const float* __restrict__ b3g,
                       float* __restrict__ uhat) {
  // Arena, phase-overlaid: xs(f32, stage1) -> w2(bf16, stage2) -> w3+b3 (stage3)
  __shared__ __align__(16) unsigned char regA[20992];
  __shared__ __align__(16) __bf16 h1l[324 * 32];   // [pix][c]   20736 B
  __shared__ __align__(16) __bf16 o2l[256 * 32];   // [n][c]     16384 B
  __shared__ __align__(16) float  w1l[16 * 32];    // [k][c] transposed
  __shared__ float b1l[32];
  __shared__ float b2l[32];

  const int tid  = threadIdx.x;
  const int bI   = blockIdx.z;
  const int g    = blockIdx.y;
  const int tile = blockIdx.x;
  const int oy = (tile >> 2) << 4, ox = (tile & 3) << 4;

  // hint weight streams into cache (global_prefetch_b8)
  __builtin_prefetch(&w2g[((size_t)(g * 32)) * 32 * 9], 0, 3);
  __builtin_prefetch(&w3g[((size_t)(g * 128)) * 32], 0, 3);

  // ---------- load relu(x) halo tile (18x18) + small weights ----------
  float* xs = (float*)regA;                               // [16][324 pad 328] f32
  for (int idx = tid; idx < 16 * 324; idx += 256) {
    int k = idx / 324, p = idx - k * 324;
    int py = p / 18, px = p - py * 18;
    int gy = oy + py - 1, gx = ox + px - 1;
    float v = 0.0f;
    if (gy >= 0 && gy < HH && gx >= 0 && gx < WW)
      v = fmaxf(xg[(((size_t)bI * 128 + g * 16 + k) * HH + gy) * WW + gx], 0.0f);
    xs[k * 328 + p] = v;
  }
  for (int idx = tid; idx < 32 * 16; idx += 256) {
    int m = idx >> 4, k = idx & 15;
    w1l[k * 32 + m] = w1g[(size_t)(g * 32 + m) * 16 + k];
  }
  if (tid < 32) { b1l[tid] = b1g[g * 32 + tid]; b2l[tid] = b2g[g * 32 + tid]; }
  __syncthreads();

  // ---------- stage 1: h1 = relu(1x1 conv); zero outside image (SAME pad) ----------
  for (int p = tid; p < 324; p += 256) {
    const int py = p / 18, px = p - py * 18;
    const int gy = oy + py - 1, gx = ox + px - 1;
    float acc[32];
    if (gy >= 0 && gy < HH && gx >= 0 && gx < WW) {
      #pragma unroll
      for (int c = 0; c < 32; ++c) acc[c] = b1l[c];
      #pragma unroll
      for (int k = 0; k < 16; ++k) {
        const float xv = xs[k * 328 + p];
        #pragma unroll
        for (int c = 0; c < 32; ++c) acc[c] += w1l[k * 32 + c] * xv;
      }
      #pragma unroll
      for (int c = 0; c < 32; ++c) acc[c] = fmaxf(acc[c], 0.0f);
    } else {
      #pragma unroll
      for (int c = 0; c < 32; ++c) acc[c] = 0.0f;
    }
    #pragma unroll
    for (int qc = 0; qc < 4; ++qc) {
      Chunk8 ch;
      #pragma unroll
      for (int j = 0; j < 8; ++j) ch.h[j] = (__bf16)acc[qc * 8 + j];
      *(v8bf*)(h1l + p * 32 + qc * 8) = ch.v;
    }
  }
  __syncthreads();

  // ---------- load w2 (bf16) into arena; K index = (dy*3+dx)*32 + c ----------
  __bf16* w2l = (__bf16*)regA;                            // [32][288 pad 296]
  for (int idx = tid; idx < 32 * 288; idx += 256) {
    int m = idx / 288, r = idx - m * 288;
    int t = r >> 5, c = r & 31;
    w2l[m * 296 + r] = (__bf16)w2g[(((size_t)(g * 32 + m)) * 32 + c) * 9 + t];
  }
  __syncthreads();

  // ---------- stage 2: 3x3 conv as WMMA GEMM  M=32, N=256, K=288 ----------
  const int wave = tid >> 5, lane = tid & 31, hi = lane >> 4, ln = lane & 15;
  for (int t2 = 0; t2 < 2; ++t2) {
    const int nt = wave + t2 * 8;                 // output row py within tile
    FAcc acc0, acc1;
    #pragma unroll
    for (int r = 0; r < 8; ++r) { acc0.f[r] = 0.0f; acc1.f[r] = 0.0f; }
    for (int kk = 0; kk < 9; ++kk) {
      const int dy = kk / 3, dx = kk - dy * 3;
      const int pix = (nt + dy) * 18 + (ln + dx);
      Frag16 bf;                                  // B: K halves hi*8+e, 16+hi*8+e = channel
      bf.h8[0] = *(const v8bf*)(h1l + pix * 32 + hi * 8);
      bf.h8[1] = *(const v8bf*)(h1l + pix * 32 + 16 + hi * 8);
      Frag16 a0, a1;                              // A: K halves hi*8+e, 16+hi*8+e
      a0.h8[0] = *(const v8bf*)(w2l + ln * 296 + kk * 32 + hi * 8);
      a0.h8[1] = *(const v8bf*)(w2l + ln * 296 + kk * 32 + 16 + hi * 8);
      a1.h8[0] = *(const v8bf*)(w2l + (16 + ln) * 296 + kk * 32 + hi * 8);
      a1.h8[1] = *(const v8bf*)(w2l + (16 + ln) * 296 + kk * 32 + 16 + hi * 8);
      acc0.v = __builtin_amdgcn_wmma_f32_16x16x32_bf16(false, a0.v, false, bf.v,
                                                       (short)0, acc0.v, false, false);
      acc1.v = __builtin_amdgcn_wmma_f32_16x16x32_bf16(false, a1.v, false, bf.v,
                                                       (short)0, acc1.v, false, false);
    }
    // D: VGPR r -> M = r + 8*hi; store transposed o2[n][c] as 16B chunks
    const int n = nt * 16 + ln;
    Chunk8 s0, s1;
    #pragma unroll
    for (int r = 0; r < 8; ++r) {
      const int m0 = r + 8 * hi, m1 = 16 + m0;
      s0.h[r] = (__bf16)fmaxf(acc0.f[r] + b2l[m0], 0.0f);
      s1.h[r] = (__bf16)fmaxf(acc1.f[r] + b2l[m1], 0.0f);
    }
    *(v8bf*)(o2l + n * 32 + 8 * hi)      = s0.v;
    *(v8bf*)(o2l + n * 32 + 16 + 8 * hi) = s1.v;
  }
  __syncthreads();

  // ---------- load w3 (bf16, pitch 40) + b3 into arena ----------
  __bf16* w3l = (__bf16*)regA;                            // [128][32 pad 40] = 10240 B
  float* b3l = (float*)(regA + 10240);
  for (int idx = tid; idx < 128 * 32; idx += 256) {
    int m = idx >> 5, c = idx & 31;
    w3l[m * 40 + c] = (__bf16)w3g[((size_t)(g * 128 + m)) * 32 + c];
  }
  if (tid < 128) b3l[tid] = b3g[g * 128 + tid];
  __syncthreads();

  // ---------- stage 3: 1x1 conv as WMMA GEMM  M=128, N=256, K=32 -> u_hat ----------
  const int mt = wave;                            // wave w owns output rows [16w,16w+16)
  for (int nt = 0; nt < 16; ++nt) {
    Frag16 bf, af;
    bf.h8[0] = *(const v8bf*)(o2l + (nt * 16 + ln) * 32 + hi * 8);
    bf.h8[1] = *(const v8bf*)(o2l + (nt * 16 + ln) * 32 + 16 + hi * 8);
    af.h8[0] = *(const v8bf*)(w3l + (mt * 16 + ln) * 40 + hi * 8);
    af.h8[1] = *(const v8bf*)(w3l + (mt * 16 + ln) * 40 + 16 + hi * 8);
    FAcc acc;
    #pragma unroll
    for (int r = 0; r < 8; ++r) acc.f[r] = 0.0f;
    acc.v = __builtin_amdgcn_wmma_f32_16x16x32_bf16(false, af.v, false, bf.v,
                                                    (short)0, acc.v, false, false);
    const int pos = (oy + nt) * WW + (ox + ln);
    float* dst = uhat + (((size_t)bI * NPOS + pos) * ICAP + g) * 128;
    const int mb = mt * 16 + 8 * hi;              // 8 contiguous output channels
    float4 lo, hip4;
    lo.x = acc.f[0] + b3l[mb + 0]; lo.y = acc.f[1] + b3l[mb + 1];
    lo.z = acc.f[2] + b3l[mb + 2]; lo.w = acc.f[3] + b3l[mb + 3];
    hip4.x = acc.f[4] + b3l[mb + 4]; hip4.y = acc.f[5] + b3l[mb + 5];
    hip4.z = acc.f[6] + b3l[mb + 6]; hip4.w = acc.f[7] + b3l[mb + 7];
    *(float4*)(dst + mb)     = lo;
    *(float4*)(dst + mb + 4) = hip4;
  }
}

// =====================================================================
// Kernel 2: dynamic routing, one wave32 per spatial position.
// lane = oc*4 + q; lane owns od in [4q, 4q+4) for all ic.
// OD reductions: lane-local (4) + shfl_xor(1)/shfl_xor(2) within the quad.
// Writes s [b][oc*16+od][pos].
// =====================================================================
__global__ __launch_bounds__(256)
void routing_kernel(const float* __restrict__ uhat, float* __restrict__ sbuf) {
  const int wave = threadIdx.x >> 5, lane = threadIdx.x & 31;
  const int pidx = blockIdx.x * 8 + wave;              // 0..32767
  const int b = pidx >> 12, pos = pidx & (NPOS - 1);
  const int oc = lane >> 2, q = lane & 3;

  float u[ICAP][4], us[ICAP][4];
  const float* base = uhat + ((size_t)(b * NPOS + pos) * ICAP) * 128 + oc * 16 + q * 4;
  #pragma unroll
  for (int ic = 0; ic < ICAP; ++ic) {
    const float4 v4 = *(const float4*)(base + ic * 128);
    u[ic][0] = v4.x; u[ic][1] = v4.y; u[ic][2] = v4.z; u[ic][3] = v4.w;
  }

  // u_sq = squash(u_hat, axis=OD)
  #pragma unroll
  for (int ic = 0; ic < ICAP; ++ic) {
    float sq = u[ic][0]*u[ic][0] + u[ic][1]*u[ic][1] + u[ic][2]*u[ic][2] + u[ic][3]*u[ic][3];
    sq += __shfl_xor(sq, 1, 32);
    sq += __shfl_xor(sq, 2, 32);
    const float fac = sq / ((0.5f + sq) * (sqrtf(sq + 1e-6f) + 1e-6f));
    #pragma unroll
    for (int j = 0; j < 4; ++j) us[ic][j] = u[ic][j] * fac;
  }

  float br[ICAP];
  #pragma unroll
  for (int ic = 0; ic < ICAP; ++ic) br[ic] = 0.0f;

  for (int it = 0; it < ROUT_ITERS - 1; ++it) {
    float s[4] = {0.f, 0.f, 0.f, 0.f};
    #pragma unroll
    for (int ic = 0; ic < ICAP; ++ic) {
      const float c = sigm(br[ic]);
      #pragma unroll
      for (int j = 0; j < 4; ++j) s[j] += us[ic][j] * c;
    }
    float ssq = s[0]*s[0] + s[1]*s[1] + s[2]*s[2] + s[3]*s[3];
    ssq += __shfl_xor(ssq, 1, 32);
    ssq += __shfl_xor(ssq, 2, 32);
    const float fac = ssq / ((0.5f + ssq) * (sqrtf(ssq + 1e-6f) + 1e-6f));
    float v[4];
    #pragma unroll
    for (int j = 0; j < 4; ++j) v[j] = s[j] * fac;
    #pragma unroll
    for (int ic = 0; ic < ICAP; ++ic) {
      float d = us[ic][0]*v[0] + us[ic][1]*v[1] + us[ic][2]*v[2] + us[ic][3]*v[3];
      d += __shfl_xor(d, 1, 32);
      d += __shfl_xor(d, 2, 32);
      br[ic] += d;
    }
  }

  // final s uses raw u_hat
  float sf[4] = {0.f, 0.f, 0.f, 0.f};
  #pragma unroll
  for (int ic = 0; ic < ICAP; ++ic) {
    const float c = sigm(br[ic]);
    #pragma unroll
    for (int j = 0; j < 4; ++j) sf[j] += u[ic][j] * c;
  }
  #pragma unroll
  for (int j = 0; j < 4; ++j)
    sbuf[((size_t)(b * OCAP + oc) * ODIM + q * 4 + j) * NPOS + pos] = sf[j];
}

// =====================================================================
// Kernel 3: spatial capsule attention + residual. One block per (b, oc).
// =====================================================================
__global__ __launch_bounds__(256)
void attn_out_kernel(const float* __restrict__ sbuf, const float* __restrict__ xg,
                     const float* __restrict__ attn_w, const float* __restrict__ attn_b,
                     float* __restrict__ out) {
  __shared__ float avg[NPOS];          // 16 KB
  __shared__ float red[256];
  __shared__ float meanhw[ODIM];
  __shared__ float stats[2];

  const int b = blockIdx.x >> 3, oc = blockIdx.x & 7;
  const int tid = threadIdx.x;
  const float* sp = sbuf + (size_t)(b * 128 + oc * 16) * NPOS;   // [od][pos]

  // pass 1: mean over H,W per od
  {
    const int od = tid >> 4, l16 = tid & 15;
    float part = 0.0f;
    for (int pos = l16; pos < NPOS; pos += 16) part += sp[od * NPOS + pos];
    red[tid] = part;
  }
  __syncthreads();
  if (tid < ODIM) {
    float s = 0.0f;
    #pragma unroll
    for (int k = 0; k < 16; ++k) s += red[tid * 16 + k];
    meanhw[tid] = s * (1.0f / (float)NPOS);
  }
  __syncthreads();

  // pass 2: avg[pos] = sum_od s*mean_hw ; accumulate sum & sumsq
  float sa = 0.0f, sa2 = 0.0f;
  for (int pos = tid; pos < NPOS; pos += 256) {
    float a = 0.0f;
    #pragma unroll
    for (int od = 0; od < ODIM; ++od) a += sp[od * NPOS + pos] * meanhw[od];
    avg[pos] = a; sa += a; sa2 += a * a;
  }
  red[tid] = sa; __syncthreads();
  for (int off = 128; off > 0; off >>= 1) { if (tid < off) red[tid] += red[tid + off]; __syncthreads(); }
  if (tid == 0) stats[0] = red[0];
  __syncthreads();
  red[tid] = sa2; __syncthreads();
  for (int off = 128; off > 0; off >>= 1) { if (tid < off) red[tid] += red[tid + off]; __syncthreads(); }
  if (tid == 0) stats[1] = red[0];
  __syncthreads();

  const float S = stats[0], S2 = stats[1];
  const float m  = S * (1.0f / (float)NPOS);
  const float var = (S2 - (float)NPOS * m * m) * (1.0f / (float)(NPOS - 1));
  const float sd = sqrtf(fmaxf(var, 0.0f)) + 1e-6f;
  const float aw = attn_w[oc], ab = attn_b[oc];

  const float* xp = xg  + (size_t)(b * 128 + oc * 16) * NPOS;
  float*       op = out + (size_t)(b * 128 + oc * 16) * NPOS;

  // pass 3: gate + residual
  for (int pos = tid; pos < NPOS; pos += 256) {
    const float tval = (avg[pos] - m) / sd * aw + ab;
    const float gate = sigm(tval);
    #pragma unroll
    for (int od = 0; od < ODIM; ++od)
      op[od * NPOS + pos] = sp[od * NPOS + pos] * gate + xp[od * NPOS + pos];
  }
}

// =====================================================================
extern "C" void kernel_launch(void* const* d_in, const int* in_sizes, int n_in,
                              void* d_out, int out_size, void* d_ws, size_t ws_size,
                              hipStream_t stream) {
  const float* x      = (const float*)d_in[0];
  const float* w1     = (const float*)d_in[1];
  const float* b1     = (const float*)d_in[2];
  const float* w2     = (const float*)d_in[3];
  const float* b2     = (const float*)d_in[4];
  const float* w3     = (const float*)d_in[5];
  const float* b3     = (const float*)d_in[6];
  const float* attn_w = (const float*)d_in[7];
  const float* attn_b = (const float*)d_in[8];

  float* uhat = (float*)d_ws;                              // [B][pos][IC][128]  = 134 MB
  float* sbuf = uhat + (size_t)BATCH * NPOS * ICAP * 128;  // [B][128][pos]      = 16.8 MB

  conv_stack_kernel<<<dim3(16, ICAP, BATCH), 256, 0, stream>>>(
      x, w1, b1, w2, b2, w3, b3, uhat);
  routing_kernel<<<dim3((BATCH * NPOS) / 8), 256, 0, stream>>>(uhat, sbuf);
  attn_out_kernel<<<dim3(BATCH * OCAP), 256, 0, stream>>>(
      sbuf, x, attn_w, attn_b, (float*)d_out);
}
